// GNoMEBlock_85031762526565
// MI455X (gfx1250) — matrix operand order repaired and checked by reading, not compiled, not run on hardware
//
#include <hip/hip_runtime.h>
#include <hip/hip_bf16.h>

#define NN 30000
#define EE 480000
#define GG 64
#define HH 128

typedef __attribute__((ext_vector_type(16))) __bf16 v16bf;
typedef __attribute__((ext_vector_type(8)))  __bf16 v8bf;
typedef __attribute__((ext_vector_type(8)))  float  v8f;

// ---------------------------------------------------------------------------
// Helpers
// ---------------------------------------------------------------------------

// Build a 16x32 bf16 A-fragment for v_wmma_f32_16x16x32_bf16.
// ISA layout (CDNA5 ISA): lane L holds row M = L%16, half = L/16.
// Element e (0..15): K = 16*(e>>3) + 8*half + (e&7)  (relative to K-tile base).
// => two runs of 8 consecutive K values at (koff + 8*half) and (koff + 16 + 8*half).
__device__ __forceinline__ v16bf load_a_global(const float* __restrict__ rowptr,
                                               int koff, int half, float scale) {
  const float* p = rowptr + koff + 8 * half;
  float4 a0 = ((const float4*)p)[0];
  float4 a1 = ((const float4*)p)[1];
  const float* q = p + 16;
  float4 a2 = ((const float4*)q)[0];
  float4 a3 = ((const float4*)q)[1];
  v16bf a;
  a[0]  = (__bf16)(a0.x * scale); a[1]  = (__bf16)(a0.y * scale);
  a[2]  = (__bf16)(a0.z * scale); a[3]  = (__bf16)(a0.w * scale);
  a[4]  = (__bf16)(a1.x * scale); a[5]  = (__bf16)(a1.y * scale);
  a[6]  = (__bf16)(a1.z * scale); a[7]  = (__bf16)(a1.w * scale);
  a[8]  = (__bf16)(a2.x * scale); a[9]  = (__bf16)(a2.y * scale);
  a[10] = (__bf16)(a2.z * scale); a[11] = (__bf16)(a2.w * scale);
  a[12] = (__bf16)(a3.x * scale); a[13] = (__bf16)(a3.y * scale);
  a[14] = (__bf16)(a3.z * scale); a[15] = (__bf16)(a3.w * scale);
  return a;
}

// A-fragment from a bf16 row staged in LDS (hidden activations, row = M).
__device__ __forceinline__ v16bf load_a_lds(const __bf16* hrow, int kb, int half) {
  const __bf16* p = hrow + kb + 8 * half;
  v8bf x0 = *(const v8bf*)p;
  v8bf x1 = *(const v8bf*)(p + 16);
  v16bf a;
#pragma unroll
  for (int i = 0; i < 8; i++) { a[i] = x0[i]; a[8 + i] = x1[i]; }
  return a;
}

__device__ __forceinline__ float silu(float v) {
  return v / (1.0f + __expf(-v));
}

// ---------------------------------------------------------------------------
// Weight pre-pack: fp32 W[K, Ncols] (row-major) -> bf16 B-fragment order.
// B 32x16 layout: lane L holds col N = L%16, half = L/16, element e: K = 16*(e>>3) + 8*half + (e&7).
// packed[((ntile*nkk + kk)*32 + lane)*16 + e]
// ---------------------------------------------------------------------------
__global__ void pack_w(const float* __restrict__ W, __bf16* __restrict__ out,
                       int K, int Ncols) {
  int idx = blockIdx.x * 256 + threadIdx.x;
  int nkk = K >> 5;
  int total = (Ncols >> 4) * nkk * 512;
  if (idx >= total) return;
  int e     = idx & 15;
  int lane  = (idx >> 4) & 31;
  int t     = idx >> 9;
  int kk    = t % nkk;
  int ntile = t / nkk;
  int Kidx  = kk * 32 + ((e >> 3) << 4) + ((lane >> 4) << 3) + (e & 7);
  int n     = ntile * 16 + (lane & 15);
  out[idx] = (__bf16)W[(size_t)Kidx * Ncols + n];
}

__global__ void zero_f32(float* __restrict__ p, int n) {
  int i = blockIdx.x * 256 + threadIdx.x;
  if (i < n) p[i] = 0.0f;
}

__global__ void count_nodes(const int* __restrict__ batch, float* __restrict__ counts) {
  int i = blockIdx.x * 256 + threadIdx.x;
  if (i < NN) atomicAdd(&counts[batch[i]], 1.0f);
}

// ---------------------------------------------------------------------------
// Edge MLP: [E,3H] -> silu -> [E,H]; scatter-add messages/16 into dst nodes.
// 1 wave per 16-edge tile; 4 waves per block; 30000 tiles exactly.
// ---------------------------------------------------------------------------
__global__ __launch_bounds__(128) void edge_mlp_kernel(
    const float* __restrict__ x, const int* __restrict__ srcI,
    const int* __restrict__ dstI, const float* __restrict__ eattr,
    const __bf16* __restrict__ w1p, const float* __restrict__ b1,
    const __bf16* __restrict__ w2p, const float* __restrict__ b2,
    float* __restrict__ out_edge, float* __restrict__ edge_msg) {
  __shared__ __attribute__((aligned(16))) __bf16 hid[4][16 * HH];
  int lane = threadIdx.x & 31;
  int wave = threadIdx.x >> 5;
  int tile = blockIdx.x * 4 + wave;
  int half = lane >> 4;
  int row  = lane & 15;
  int e_base = tile * 16;
  int eL = e_base + row;

  const float* rp0 = x + (size_t)srcI[eL] * HH;
  const float* rp1 = x + (size_t)dstI[eL] * HH;
  const float* rp2 = eattr + (size_t)eL * HH;

  v8f acc[8] = {};
#pragma unroll
  for (int kk = 0; kk < 12; kk++) {
    int kb = kk * 32;
    const float* rp = (kb < 128) ? rp0 : ((kb < 256) ? rp1 : rp2);
    v16bf a = load_a_global(rp, kb & 127, half, 1.0f);
#pragma unroll
    for (int n = 0; n < 8; n++) {
      v16bf b = *(const v16bf*)(w1p + ((size_t)(n * 12 + kk) * 32 + lane) * 16);
      acc[n] = __builtin_amdgcn_wmma_f32_16x16x32_bf16(false, a, false, b,
                                                       (short)0, acc[n], false, false);
    }
  }

  // bias + SiLU -> LDS (D layout: lane col h = 16n+row, rows M = r + 8*half)
  __bf16* ht = hid[wave];
#pragma unroll
  for (int n = 0; n < 8; n++) {
    int h = 16 * n + row;
    float bb = b1[h];
#pragma unroll
    for (int r = 0; r < 8; r++) {
      ht[(r + 8 * half) * HH + h] = (__bf16)silu(acc[n][r] + bb);
    }
  }
  __syncthreads();

  v8f acc2[8] = {};
  const __bf16* hrow = ht + row * HH;
#pragma unroll
  for (int kk = 0; kk < 4; kk++) {
    v16bf a = load_a_lds(hrow, kk * 32, half);
#pragma unroll
    for (int n = 0; n < 8; n++) {
      v16bf b = *(const v16bf*)(w2p + ((size_t)(n * 4 + kk) * 32 + lane) * 16);
      acc2[n] = __builtin_amdgcn_wmma_f32_16x16x32_bf16(false, a, false, b,
                                                        (short)0, acc2[n], false, false);
    }
  }

  int drow[8];
#pragma unroll
  for (int r = 0; r < 8; r++) drow[r] = dstI[e_base + r + 8 * half];
#pragma unroll
  for (int n = 0; n < 8; n++) {
    int h = 16 * n + row;
    float bb = b2[h];
#pragma unroll
    for (int r = 0; r < 8; r++) {
      float v = acc2[n][r] + bb;
      out_edge[(size_t)(e_base + r + 8 * half) * HH + h] = v;
      atomicAdd(&edge_msg[(size_t)drow[r] * HH + h], v * 0.0625f);
    }
  }
}

// ---------------------------------------------------------------------------
// Node MLP: [N,3H]=[x, msgs, u[batch]] -> x_new; accumulate per-graph sums.
// 1 wave per 16-node tile; 5 waves/block; 1875 tiles = 375 blocks exactly.
// ---------------------------------------------------------------------------
__global__ __launch_bounds__(160) void node_mlp_kernel(
    const float* __restrict__ x, const float* __restrict__ edge_msg,
    const float* __restrict__ u, const int* __restrict__ batch,
    const __bf16* __restrict__ w1p, const float* __restrict__ b1,
    const __bf16* __restrict__ w2p, const float* __restrict__ b2,
    float* __restrict__ out_node, float* __restrict__ graph_sum) {
  __shared__ __attribute__((aligned(16))) __bf16 hid[5][16 * HH];
  int lane = threadIdx.x & 31;
  int wave = threadIdx.x >> 5;
  int tile = blockIdx.x * 5 + wave;
  int half = lane >> 4;
  int row  = lane & 15;
  int n_base = tile * 16;
  int nL = n_base + row;

  const float* rp0 = x + (size_t)nL * HH;
  const float* rp1 = edge_msg + (size_t)nL * HH;   // already holds sum; scale by 1/16 on load
  const float* rp2 = u + (size_t)batch[nL] * HH;

  v8f acc[8] = {};
#pragma unroll
  for (int kk = 0; kk < 12; kk++) {
    int kb = kk * 32;
    const float* rp = (kb < 128) ? rp0 : ((kb < 256) ? rp1 : rp2);
    float sc = (kb >= 128 && kb < 256) ? 0.0625f : 1.0f;
    v16bf a = load_a_global(rp, kb & 127, half, sc);
#pragma unroll
    for (int n = 0; n < 8; n++) {
      v16bf b = *(const v16bf*)(w1p + ((size_t)(n * 12 + kk) * 32 + lane) * 16);
      acc[n] = __builtin_amdgcn_wmma_f32_16x16x32_bf16(false, a, false, b,
                                                       (short)0, acc[n], false, false);
    }
  }

  __bf16* ht = hid[wave];
#pragma unroll
  for (int n = 0; n < 8; n++) {
    int h = 16 * n + row;
    float bb = b1[h];
#pragma unroll
    for (int r = 0; r < 8; r++) {
      ht[(r + 8 * half) * HH + h] = (__bf16)silu(acc[n][r] + bb);
    }
  }
  __syncthreads();

  v8f acc2[8] = {};
  const __bf16* hrow = ht + row * HH;
#pragma unroll
  for (int kk = 0; kk < 4; kk++) {
    v16bf a = load_a_lds(hrow, kk * 32, half);
#pragma unroll
    for (int n = 0; n < 8; n++) {
      v16bf b = *(const v16bf*)(w2p + ((size_t)(n * 4 + kk) * 32 + lane) * 16);
      acc2[n] = __builtin_amdgcn_wmma_f32_16x16x32_bf16(false, a, false, b,
                                                        (short)0, acc2[n], false, false);
    }
  }

  int grow[8];
#pragma unroll
  for (int r = 0; r < 8; r++) grow[r] = batch[n_base + r + 8 * half];
#pragma unroll
  for (int n = 0; n < 8; n++) {
    int h = 16 * n + row;
    float bb = b2[h];
#pragma unroll
    for (int r = 0; r < 8; r++) {
      float v = acc2[n][r] + bb;
      out_node[(size_t)(n_base + r + 8 * half) * HH + h] = v;
      atomicAdd(&graph_sum[(size_t)grow[r] * HH + h], v);
    }
  }
}

// ---------------------------------------------------------------------------
// Global MLP: [G,2H]=[u, mean(x_new per graph)] -> u_new. One block, 4 waves.
// ---------------------------------------------------------------------------
__global__ __launch_bounds__(128) void global_mlp_kernel(
    const float* __restrict__ u, const float* __restrict__ gsum,
    const float* __restrict__ counts,
    const __bf16* __restrict__ w1p, const float* __restrict__ b1,
    const __bf16* __restrict__ w2p, const float* __restrict__ b2,
    float* __restrict__ out_u) {
  __shared__ __attribute__((aligned(16))) __bf16 hid[4][16 * HH];
  int lane = threadIdx.x & 31;
  int wave = threadIdx.x >> 5;
  int half = lane >> 4;
  int row  = lane & 15;
  int gr = wave * 16 + row;

  const float* rp0 = u + (size_t)gr * HH;
  const float* rp1 = gsum + (size_t)gr * HH;
  float invc = 1.0f / fmaxf(counts[gr], 1.0f);

  v8f acc[8] = {};
#pragma unroll
  for (int kk = 0; kk < 8; kk++) {
    const float* rp = (kk < 4) ? rp0 : rp1;
    float sc = (kk < 4) ? 1.0f : invc;
    v16bf a = load_a_global(rp, (kk & 3) * 32, half, sc);
#pragma unroll
    for (int n = 0; n < 8; n++) {
      v16bf b = *(const v16bf*)(w1p + ((size_t)(n * 8 + kk) * 32 + lane) * 16);
      acc[n] = __builtin_amdgcn_wmma_f32_16x16x32_bf16(false, a, false, b,
                                                       (short)0, acc[n], false, false);
    }
  }

  __bf16* ht = hid[wave];
#pragma unroll
  for (int n = 0; n < 8; n++) {
    int h = 16 * n + row;
    float bb = b1[h];
#pragma unroll
    for (int r = 0; r < 8; r++) {
      ht[(r + 8 * half) * HH + h] = (__bf16)silu(acc[n][r] + bb);
    }
  }
  __syncthreads();

  v8f acc2[8] = {};
  const __bf16* hrow = ht + row * HH;
#pragma unroll
  for (int kk = 0; kk < 4; kk++) {
    v16bf a = load_a_lds(hrow, kk * 32, half);
#pragma unroll
    for (int n = 0; n < 8; n++) {
      v16bf b = *(const v16bf*)(w2p + ((size_t)(n * 4 + kk) * 32 + lane) * 16);
      acc2[n] = __builtin_amdgcn_wmma_f32_16x16x32_bf16(false, a, false, b,
                                                        (short)0, acc2[n], false, false);
    }
  }

#pragma unroll
  for (int n = 0; n < 8; n++) {
    int h = 16 * n + row;
    float bb = b2[h];
#pragma unroll
    for (int r = 0; r < 8; r++) {
      out_u[(size_t)(wave * 16 + r + 8 * half) * HH + h] = acc2[n][r] + bb;
    }
  }
}

// ---------------------------------------------------------------------------
// Launch
// ---------------------------------------------------------------------------
extern "C" void kernel_launch(void* const* d_in, const int* in_sizes, int n_in,
                              void* d_out, int out_size, void* d_ws, size_t ws_size,
                              hipStream_t stream) {
  const float* x         = (const float*)d_in[0];
  const int*   edge_idx  = (const int*)d_in[1];   // (2,E): src then dst
  const float* edge_attr = (const float*)d_in[2];
  const float* u         = (const float*)d_in[3];
  const int*   batch     = (const int*)d_in[4];
  const float* eW1 = (const float*)d_in[5];  const float* eb1 = (const float*)d_in[6];
  const float* eW2 = (const float*)d_in[7];  const float* eb2 = (const float*)d_in[8];
  const float* nW1 = (const float*)d_in[9];  const float* nb1 = (const float*)d_in[10];
  const float* nW2 = (const float*)d_in[11]; const float* nb2 = (const float*)d_in[12];
  const float* gW1 = (const float*)d_in[13]; const float* gb1 = (const float*)d_in[14];
  const float* gW2 = (const float*)d_in[15]; const float* gb2 = (const float*)d_in[16];

  float* out      = (float*)d_out;
  float* out_node = out;                            // N*H
  float* out_edge = out + (size_t)NN * HH;          // E*H
  float* out_u    = out + (size_t)(NN + EE) * HH;   // G*H

  // Workspace layout
  float* edge_msg  = (float*)d_ws;                  // N*H
  float* graph_sum = edge_msg + (size_t)NN * HH;    // G*H
  float* counts    = graph_sum + (size_t)GG * HH;   // G
  __bf16* eW1p = (__bf16*)(counts + GG);            // 8*12*512
  __bf16* eW2p = eW1p + 8 * 12 * 512;               // 8*4*512
  __bf16* nW1p = eW2p + 8 * 4 * 512;
  __bf16* nW2p = nW1p + 8 * 12 * 512;
  __bf16* gW1p = nW2p + 8 * 4 * 512;                // 8*8*512
  __bf16* gW2p = gW1p + 8 * 8 * 512;

  // 1. zero accumulators (edge_msg, graph_sum, counts are contiguous)
  int zc = NN * HH + GG * HH + GG;
  zero_f32<<<(zc + 255) / 256, 256, 0, stream>>>(edge_msg, zc);

  // 2. pack weights to bf16 B-fragment order
  pack_w<<<(8 * 12 * 512 + 255) / 256, 256, 0, stream>>>(eW1, eW1p, 3 * HH, HH);
  pack_w<<<(8 * 4 * 512 + 255) / 256, 256, 0, stream>>>(eW2, eW2p, HH, HH);
  pack_w<<<(8 * 12 * 512 + 255) / 256, 256, 0, stream>>>(nW1, nW1p, 3 * HH, HH);
  pack_w<<<(8 * 4 * 512 + 255) / 256, 256, 0, stream>>>(nW2, nW2p, HH, HH);
  pack_w<<<(8 * 8 * 512 + 255) / 256, 256, 0, stream>>>(gW1, gW1p, 2 * HH, HH);
  pack_w<<<(8 * 4 * 512 + 255) / 256, 256, 0, stream>>>(gW2, gW2p, HH, HH);

  // 3. node counts per graph
  count_nodes<<<(NN + 255) / 256, 256, 0, stream>>>(batch, counts);

  // 4. edge MLP + message scatter (30000 tiles, 4 waves/block)
  edge_mlp_kernel<<<EE / 16 / 4, 128, 0, stream>>>(
      x, edge_idx, edge_idx + EE, edge_attr, eW1p, eb1, eW2p, eb2,
      out_edge, edge_msg);

  // 5. node MLP + per-graph sum (1875 tiles, 5 waves/block)
  node_mlp_kernel<<<NN / 16 / 5, 160, 0, stream>>>(
      x, edge_msg, u, batch, nW1p, nb1, nW2p, nb2, out_node, graph_sum);

  // 6. global MLP (4 tiles, 1 block of 4 waves)
  global_mlp_kernel<<<1, 128, 0, stream>>>(
      u, graph_sum, counts, gW1p, gb1, gW2p, gb2, out_u);
}
